// Mamba2_11476152615136
// MI455X (gfx1250) — matrix operand order repaired and checked by reading, not compiled, not run on hardware
//
#include <hip/hip_runtime.h>
#include <stdint.h>

typedef __attribute__((ext_vector_type(16))) __bf16 v16bf;
typedef __attribute__((ext_vector_type(8)))  float  v8f;

#define DEV __device__ __forceinline__

DEV unsigned short f2bfbits(float f) {
  union { float f; unsigned u; } x; x.f = f;
  unsigned r = x.u + 0x7fffu + ((x.u >> 16) & 1u);   // round-to-nearest-even
  return (unsigned short)(r >> 16);
}
DEV __bf16 f2bf(float f) { return __builtin_bit_cast(__bf16, f2bfbits(f)); }
DEV float bf2f(__bf16 b) {
  unsigned u = ((unsigned)__builtin_bit_cast(unsigned short, b)) << 16;
  return __builtin_bit_cast(float, u);
}

DEV v8f wmma_bf16(v16bf a, v16bf b, v8f c) {
  return __builtin_amdgcn_wmma_f32_16x16x32_bf16(false, a, false, b, (short)0, c,
                                                 false, false);
}

// CDNA5 async global->LDS copy (16B per lane), tracked by ASYNCcnt.
DEV void async_b128(unsigned lds_off, unsigned long long gaddr) {
  asm volatile("global_load_async_to_lds_b128 %0, %1, off"
               :: "v"(lds_off), "v"(gaddr) : "memory");
}
template<int N> DEV void wait_async() {
  asm volatile("s_wait_asynccnt %0" :: "n"(N) : "memory");
}

// ---------------------------------------------------------------------------
// fp32 -> bf16 bulk convert (4 elems/thread, exact-sized grid)
// ---------------------------------------------------------------------------
__global__ __launch_bounds__(256) void cvt_kernel(const float4* __restrict__ s,
                                                  uint2* __restrict__ d) {
  size_t i = (size_t)blockIdx.x * 256 + threadIdx.x;
  float4 v = s[i];
  uint2 o;
  o.x = (unsigned)f2bfbits(v.x) | ((unsigned)f2bfbits(v.y) << 16);
  o.y = (unsigned)f2bfbits(v.z) | ((unsigned)f2bfbits(v.w) << 16);
  d[i] = o;
}

// ---------------------------------------------------------------------------
// GEMM: C[M][N] = A[M][K] * W[N][K]^T, A/W bf16 in global, f32 accumulate.
// BM=128, BN=64, BK=64. 256 threads = 8 waves (4Mx2N), wave tile 32x32.
// Double-buffered LDS fed by async global->LDS b128 copies.
// ---------------------------------------------------------------------------
template<bool OUT_F32>
__global__ __launch_bounds__(256) void gemm_async_kernel(
    const __bf16* __restrict__ Ab, const __bf16* __restrict__ Wb,
    void* __restrict__ Cptr, int N, int K)
{
  constexpr int BM = 128, BN = 64, LDC = 72;       // 144B rows (16B multiple)
  __shared__ __bf16 As[2][BM][LDC];
  __shared__ __bf16 Bs[2][BN][LDC];

  const int tid  = threadIdx.x;
  const int lane = tid & 31, half = lane >> 4, l16 = lane & 15;
  const int wave = tid >> 5, wm = wave >> 1, wn = wave & 1;
  const int m0 = blockIdx.y * BM, n0 = blockIdx.x * BN;

  auto issue = [&](int k0, int buf) {
    #pragma unroll
    for (int i = tid; i < BM * 8; i += 256) {      // A: 128 rows x 8 x16B
      int r = i >> 3, c = (i & 7) * 8;
      async_b128((unsigned)(uintptr_t)&As[buf][r][c],
                 (unsigned long long)(uintptr_t)(Ab + (size_t)(m0 + r) * K + k0 + c));
    }
    #pragma unroll
    for (int i = tid; i < BN * 8; i += 256) {      // B: 64 rows x 8 x16B
      int r = i >> 3, c = (i & 7) * 8;
      async_b128((unsigned)(uintptr_t)&Bs[buf][r][c],
                 (unsigned long long)(uintptr_t)(Wb + (size_t)(n0 + r) * K + k0 + c));
    }
  };  // 6 async instructions per wave per tile

  const v8f vzero = {0.f,0.f,0.f,0.f,0.f,0.f,0.f,0.f};
  v8f acc[2][2];
  acc[0][0] = vzero; acc[0][1] = vzero; acc[1][0] = vzero; acc[1][1] = vzero;

  const int nk = K / 64;
  issue(0, 0);
  for (int t = 0; t < nk; ++t) {
    if (t + 1 < nk) { issue((t + 1) * 64, (t + 1) & 1); wait_async<6>(); }
    else            { wait_async<0>(); }
    __syncthreads();
    const int buf = t & 1;

    #pragma unroll
    for (int kf = 0; kf < 2; ++kf) {
      v16bf a[2], b[2];
      #pragma unroll
      for (int im = 0; im < 2; ++im) {
        int mr = wm * 32 + im * 16 + l16;
        #pragma unroll
        for (int e = 0; e < 16; ++e) {
          int kk = kf * 32 + (e >> 3) * 16 + half * 8 + (e & 7);  // A layout
          a[im][e] = As[buf][mr][kk];
        }
      }
      #pragma unroll
      for (int in_ = 0; in_ < 2; ++in_) {
        int nr = wn * 32 + in_ * 16 + l16;
        #pragma unroll
        for (int e = 0; e < 16; ++e)
          b[in_][e] = Bs[buf][nr][kf * 32 + (e >> 3) * 16 + half * 8 + (e & 7)];        // B layout
      }
      #pragma unroll
      for (int im = 0; im < 2; ++im)
        #pragma unroll
        for (int in_ = 0; in_ < 2; ++in_)
          acc[im][in_] = wmma_bf16(a[im], b[in_], acc[im][in_]);
    }
    __syncthreads();
  }

  // C layout: lane holds column l16, rows r + 8*half
  #pragma unroll
  for (int im = 0; im < 2; ++im)
    #pragma unroll
    for (int in_ = 0; in_ < 2; ++in_)
      #pragma unroll
      for (int r = 0; r < 8; ++r) {
        int m = m0 + wm * 32 + im * 16 + r + 8 * half;
        int n = n0 + wn * 32 + in_ * 16 + l16;
        float val = acc[im][in_][r];
        if (OUT_F32) ((float*)Cptr)[(size_t)m * N + n] = val;
        else         ((__bf16*)Cptr)[(size_t)m * N + n] = f2bf(val);
      }
}

// ---------------------------------------------------------------------------
// RoPE + split of in-proj output Y[B*S][6144] (bf16):
//   [0,1024)->v copy, [1024,2048)->k RoPE, [2048,6144)->q RoPE
// q: [B][32][S][128], k/v: [B][8][S][128]  (bf16)
// ---------------------------------------------------------------------------
__global__ __launch_bounds__(256) void rope_split_kernel(
    const __bf16* __restrict__ Y, __bf16* __restrict__ Q,
    __bf16* __restrict__ Kq, __bf16* __restrict__ Vq)
{
  const int row = blockIdx.x;           // b*2048 + s
  const int b = row >> 11, s = row & 2047;
  const __bf16* yr = Y + (size_t)row * 6144;

  for (int idx = threadIdx.x; idx < 6144; idx += 256) {
    if (idx < 1024) {
      int c = idx, hh = c >> 7, d = c & 127;
      Vq[((size_t)(b * 8 + hh) * 2048 + s) * 128 + d] = yr[idx];
    } else {
      const bool is_k = idx < 2048;
      const int base = is_k ? 1024 : 2048;
      int c = idx - base, hh = c >> 7, d = c & 127;
      float x = bf2f(yr[idx]);
      int dp = (d < 64) ? d + 64 : d - 64;
      float xp = bf2f(yr[base + hh * 128 + dp]);
      float rot = (d < 64) ? -xp : xp;
      float inv = __powf(10000.0f, -(float)(d & 63) * (1.0f / 64.0f));
      float ang = (float)s * inv;
      float o = x * __cosf(ang) + rot * __sinf(ang);
      if (is_k) Kq[((size_t)(b * 8  + hh) * 2048 + s) * 128 + d] = f2bf(o);
      else      Q [((size_t)(b * 32 + hh) * 2048 + s) * 128 + d] = f2bf(o);
    }
  }
}

// ---------------------------------------------------------------------------
// Causal flash attention. Block = 128 queries of one (b,h); 8 waves, each
// owns 16 query rows. Key tiles of 32, double-buffered async K/V staging.
// GQA: kv head = h/4. Output O[B][S][32*128] bf16.
// ---------------------------------------------------------------------------
__global__ __launch_bounds__(256) void attn_kernel(
    const __bf16* __restrict__ Q, const __bf16* __restrict__ Kb,
    const __bf16* __restrict__ Vb, __bf16* __restrict__ O)
{
  constexpr int S = 2048, D = 128, KT = 32;
  __shared__ __bf16 Ks[2][KT][D + 8];
  __shared__ __bf16 Vs[2][KT][D + 8];
  __shared__ __bf16 Ps[8][16][48];      // per-wave P strip 16x32, 96B rows

  const int tid = threadIdx.x;
  const int w = tid >> 5, lane = tid & 31;
  const int half = lane >> 4, l16 = lane & 15;
  const int bh = blockIdx.y;            // b*32 + h
  const int b = bh >> 5, h = bh & 31;
  const int q0 = blockIdx.x * 128;
  const int kvh = h >> 2;

  const __bf16* qh = Q  + ((size_t)(b * 32 + h)   * S) * D;
  const __bf16* kh = Kb + ((size_t)(b * 8 + kvh) * S) * D;
  const __bf16* vh = Vb + ((size_t)(b * 8 + kvh) * S) * D;

  auto issue_kv = [&](int t, int buf) {
    const int n0 = t * KT;
    #pragma unroll
    for (int i = tid; i < KT * 16; i += 256) {     // 32 rows x 16 x16B
      int r = i >> 4, c = (i & 15) * 8;
      async_b128((unsigned)(uintptr_t)&Ks[buf][r][c],
                 (unsigned long long)(uintptr_t)(kh + (size_t)(n0 + r) * D + c));
      async_b128((unsigned)(uintptr_t)&Vs[buf][r][c],
                 (unsigned long long)(uintptr_t)(vh + (size_t)(n0 + r) * D + c));
    }
  };  // 4 async instructions per wave per tile

  // Q fragments (A layout), rows q0 + w*16 .. +15, K=128 -> 4 frags of 32
  const int qrow = q0 + w * 16 + l16;
  v16bf aq[4];
  #pragma unroll
  for (int f = 0; f < 4; ++f)
    #pragma unroll
    for (int e = 0; e < 16; ++e) {
      int kk = f * 32 + (e >> 3) * 16 + half * 8 + (e & 7);
      aq[f][e] = qh[(size_t)qrow * D + kk];
    }

  const v8f vzero = {0.f,0.f,0.f,0.f,0.f,0.f,0.f,0.f};
  v8f o[8];
  float rm[8], rl[8];
  #pragma unroll
  for (int i = 0; i < 8; ++i) { o[i] = vzero; rm[i] = -1e30f; rl[i] = 0.f; }

  const int ntiles = 4 * blockIdx.x + 4;           // keys up to q0+127
  const float scale = 0.08838834764831845f;        // 1/sqrt(128)

  issue_kv(0, 0);
  for (int t = 0; t < ntiles; ++t) {
    if (t + 1 < ntiles) { issue_kv(t + 1, (t + 1) & 1); wait_async<4>(); }
    else                { wait_async<0>(); }
    __syncthreads();
    const int buf = t & 1;
    const int n0 = t * KT;

    // scores = Q K^T : 2 n-subtiles x 4 K-steps
    v8f sc[2];
    sc[0] = vzero; sc[1] = vzero;
    #pragma unroll
    for (int nt = 0; nt < 2; ++nt)
      #pragma unroll
      for (int kf = 0; kf < 4; ++kf) {
        v16bf bk;
        #pragma unroll
        for (int e = 0; e < 16; ++e)
          bk[e] = Ks[buf][nt * 16 + l16][kf * 32 + (e >> 3) * 16 + half * 8 + (e & 7)];
        sc[nt] = wmma_bf16(aq[kf], bk, sc[nt]);
      }

    // online softmax (row m = r + 8*half; reduce over 16-lane half-group)
    #pragma unroll
    for (int r = 0; r < 8; ++r) {
      const int rowg = q0 + w * 16 + r + 8 * half;
      float tmax = -1e30f;
      #pragma unroll
      for (int nt = 0; nt < 2; ++nt) {
        float s_ = sc[nt][r] * scale;
        int keyg = n0 + nt * 16 + l16;
        s_ = (keyg > rowg) ? -1e30f : s_;
        sc[nt][r] = s_;
        tmax = fmaxf(tmax, s_);
      }
      #pragma unroll
      for (int mk = 1; mk < 16; mk <<= 1)
        tmax = fmaxf(tmax, __shfl_xor(tmax, mk, 32));
      float mnew = fmaxf(rm[r], tmax);
      float corr = __expf(rm[r] - mnew);
      rm[r] = mnew;
      float psum = 0.f;
      #pragma unroll
      for (int nt = 0; nt < 2; ++nt) {
        float p = __expf(sc[nt][r] - mnew);
        sc[nt][r] = p;
        psum += p;
      }
      #pragma unroll
      for (int mk = 1; mk < 16; mk <<= 1)
        psum += __shfl_xor(psum, mk, 32);
      rl[r] = rl[r] * corr + psum;
      #pragma unroll
      for (int dt = 0; dt < 8; ++dt) o[dt][r] = o[dt][r] * corr;
    }

    // P: C-layout -> A-layout via per-wave LDS strip (same-wave dependency)
    #pragma unroll
    for (int nt = 0; nt < 2; ++nt)
      #pragma unroll
      for (int r = 0; r < 8; ++r)
        Ps[w][r + 8 * half][nt * 16 + l16] = f2bf(sc[nt][r]);

    // O += P V : single K=32 step x 8 d-subtiles
    v16bf ap;
    #pragma unroll
    for (int e = 0; e < 16; ++e)
      ap[e] = Ps[w][l16][(e >> 3) * 16 + half * 8 + (e & 7)];
    #pragma unroll
    for (int dt = 0; dt < 8; ++dt) {
      v16bf bv;
      #pragma unroll
      for (int e = 0; e < 16; ++e)
        bv[e] = Vs[buf][(e >> 3) * 16 + half * 8 + (e & 7)][dt * 16 + l16];
      o[dt] = wmma_bf16(ap, bv, o[dt]);
    }
    __syncthreads();
  }

  // Normalize and write O[b][s][h*128 + d] (bf16)
  #pragma unroll
  for (int dt = 0; dt < 8; ++dt)
    #pragma unroll
    for (int r = 0; r < 8; ++r) {
      int rowg = q0 + w * 16 + r + 8 * half;
      int d = dt * 16 + l16;
      float val = o[dt][r] / rl[r];
      O[((size_t)(b * S + rowg)) * 4096 + h * 128 + d] = f2bf(val);
    }
}

// ---------------------------------------------------------------------------
// Launch
// ---------------------------------------------------------------------------
extern "C" void kernel_launch(void* const* d_in, const int* in_sizes, int n_in,
                              void* d_out, int out_size, void* d_ws, size_t ws_size,
                              hipStream_t stream) {
  (void)in_sizes; (void)n_in; (void)out_size; (void)ws_size;
  const float* u     = (const float*)d_in[0];   // [2][2048][2048]
  const float* W_in  = (const float*)d_in[1];   // [10272][2048]
  const float* W_out = (const float*)d_in[2];   // [2048][4096]
  float* out = (float*)d_out;                   // [2][2048][2048]

  char* ws = (char*)d_ws;
  // ws layout (bytes):
  //   [0, 50331648)          Y bf16 [4096][6144]  (reused as O [4096][4096])
  //   [50331648, 83886080)   q bf16 [2][32][2048][128]
  //   [83886080, 92274688)   k bf16 [2][8][2048][128]
  //   [92274688, 100663296)  v bf16
  //   [100663296,117440512)  u bf16 [4096][2048]
  //   [117440512,142606336)  W_in slice bf16 [6144][2048]
  //   [142606336,159383552)  W_out bf16 [2048][4096]
  __bf16* Y     = (__bf16*)ws;
  __bf16* Qb    = (__bf16*)(ws + 50331648u);
  __bf16* Kb    = (__bf16*)(ws + 83886080u);
  __bf16* Vb    = (__bf16*)(ws + 92274688u);
  __bf16* ub    = (__bf16*)(ws + 100663296u);
  __bf16* Winb  = (__bf16*)(ws + 117440512u);
  __bf16* Woutb = (__bf16*)(ws + 142606336u);
  __bf16* O     = (__bf16*)ws;                  // Y dead after rope_split

  // 0) one-shot fp32 -> bf16 conversions (bandwidth trivial)
  cvt_kernel<<<8192,  256, 0, stream>>>((const float4*)u, (uint2*)ub);
  cvt_kernel<<<12288, 256, 0, stream>>>((const float4*)(W_in + (size_t)4096 * 2048),
                                        (uint2*)Winb);
  cvt_kernel<<<8192,  256, 0, stream>>>((const float4*)W_out, (uint2*)Woutb);

  // 1) in-proj (only needed 6144 features)
  gemm_async_kernel<false><<<dim3(6144 / 64, 4096 / 128), 256, 0, stream>>>(
      ub, Winb, Y, 6144, 2048);

  // 2) RoPE + q/k/v split
  rope_split_kernel<<<4096, 256, 0, stream>>>(Y, Qb, Kb, Vb);

  // 3) causal flash attention
  attn_kernel<<<dim3(16, 64), 256, 0, stream>>>(Qb, Kb, Vb, O);

  // 4) out-proj -> fp32 output
  gemm_async_kernel<true><<<dim3(2048 / 64, 4096 / 128), 256, 0, stream>>>(
      O, Woutb, out, 2048, 4096);
}
